// VQEmbedding_85873576116443
// MI455X (gfx1250) — compile-verified, matrix operand order checked by reading, not run on hardware
//
#include <hip/hip_runtime.h>
#include <hip/hip_bf16.h>

// ---------------------------------------------------------------------------
// VQ codebook lookup for MI455X (gfx1250, wave32, WMMA + async global->LDS).
// score[n,k] = c_sqr[k] - 2 * dot(x_n, c_k)   (||x||^2 dropped: row-constant)
// GEMM via v_wmma_f32_16x16x32_bf16; codebook tiles double-buffered in LDS
// via global_load_async_to_lds_b128 (ASYNCcnt) and shared by all 8 waves.
// ---------------------------------------------------------------------------

typedef __attribute__((ext_vector_type(16))) __bf16 v16bf;
typedef __attribute__((ext_vector_type(8)))  __bf16 v8bf;
typedef __attribute__((ext_vector_type(8)))  float  v8f;
typedef __attribute__((ext_vector_type(4)))  int    v4i;

#define VQ_B 32
#define VQ_T 1024
#define VQ_D 256
#define VQ_K 1024
#define VQ_N (VQ_B * VQ_T)          // 32768 rows
#define ROWS_PER_WG 128             // 8 waves * 16 rows
#define KSTEPS (VQ_D / 32)          // 8 wmma k-steps over D=256
#define KTILES (VQ_K / 16)          // 64 tiles of 16 codes

#define LDS_ROW   528               // 512B of bf16 data + 16B pad (bank spread)
#define TILE_SZ   (16 * LDS_ROW)    // one staged codebook tile (8.25 KB)

#define AS1 __attribute__((address_space(1)))
#define AS3 __attribute__((address_space(3)))

#if __has_builtin(__builtin_amdgcn_global_load_async_to_lds_b128)
#define VQ_ASYNC 1
#endif

__device__ __forceinline__ void vq_wait_async() {
#if defined(VQ_ASYNC)
#if __has_builtin(__builtin_amdgcn_s_wait_asynccnt)
    __builtin_amdgcn_s_wait_asynccnt(0);
#else
    asm volatile("s_wait_asynccnt 0" ::: "memory");
#endif
#endif
}

// Cooperative fill of one 16x256 bf16 codebook tile into LDS.
// 512 x 16B chunks; 256 threads -> 2 chunks each.
__device__ __forceinline__ void vq_fill_tile(const __bf16* __restrict__ cbb,
                                             int kt, char* lbuf, int tid) {
    #pragma unroll
    for (int q = 0; q < 2; ++q) {
        const int c   = tid + q * 256;        // 0..511
        const int r   = c >> 5;               // code row 0..15
        const int col = c & 31;               // 16B column 0..31
        const __bf16* g = cbb + ((size_t)(kt * 16 + r) * VQ_D) + col * 8;
        char* l = lbuf + r * LDS_ROW + col * 16;
#if defined(VQ_ASYNC)
        __builtin_amdgcn_global_load_async_to_lds_b128(
            (AS1 v4i*)(void*)g, (AS3 v4i*)(void*)l, 0, 0);
#else
        *(int4*)l = *(const int4*)g;
#endif
    }
}

// ------------------------- prep: bf16 codebook + c_sqr ----------------------
__global__ void vq_prep_kernel(const float* __restrict__ cb,
                               __bf16* __restrict__ cbb,
                               float* __restrict__ c_sqr) {
    __shared__ float red[256];
    const int k = blockIdx.x;       // one code per block
    const int d = threadIdx.x;      // D == 256 == blockDim.x
    float v = cb[k * VQ_D + d];
    cbb[k * VQ_D + d] = (__bf16)v;
    red[d] = v * v;
    __syncthreads();
    for (int s = 128; s > 0; s >>= 1) {
        if (d < s) red[d] += red[d + s];
        __syncthreads();
    }
    if (d == 0) c_sqr[k] = red[0];
}

// ------------------------------- main kernel --------------------------------
__global__ void __launch_bounds__(256)
vq_main_kernel(const float* __restrict__ x,        // [N, D] f32
               const float* __restrict__ cb,       // [K, D] f32 (for gather)
               const __bf16* __restrict__ cbb,     // [K, D] bf16
               const float* __restrict__ c_sqr,    // [K]
               float* __restrict__ out) {          // codes|codes|idx
    __shared__ char s_tiles[2 * TILE_SZ];          // double-buffered B tiles
    __shared__ int  s_idx[ROWS_PER_WG];

    const int tid   = threadIdx.x;
    const int lane  = tid & 31;
    const int wave  = tid >> 5;                     // 0..7
    const int lhalf = lane >> 4;                    // 0 or 1
    const int l16   = lane & 15;
    const int wgrow = blockIdx.x * ROWS_PER_WG;     // first row of this WG
    const int mbase = wgrow + wave * 16;            // first row of this wave

    // ---- kick off async fill of tile 0 while we convert A fragments ----
    vq_fill_tile(cbb, 0, s_tiles, tid);

    // ---- load A fragments (16 rows x 256 d, bf16) once into registers ----
    // ISA 16-bit A layout: lanes 0-15 row=lane, K chunks {0..7, 16..23};
    // lanes 16-31 row=lane-16, K chunks {8..15, 24..31}. Two 16B chunks/step.
    v16bf afrag[KSTEPS];
    {
        const float* xrow = x + (size_t)(mbase + l16) * VQ_D;
        #pragma unroll
        for (int ks = 0; ks < KSTEPS; ++ks) {
            const int c0 = ks * 32 + lhalf * 8;
            const float4 f0 = *(const float4*)(xrow + c0);
            const float4 f1 = *(const float4*)(xrow + c0 + 4);
            const float4 g0 = *(const float4*)(xrow + c0 + 16);
            const float4 g1 = *(const float4*)(xrow + c0 + 20);
            v16bf a;
            a[0]  = (__bf16)f0.x; a[1]  = (__bf16)f0.y;
            a[2]  = (__bf16)f0.z; a[3]  = (__bf16)f0.w;
            a[4]  = (__bf16)f1.x; a[5]  = (__bf16)f1.y;
            a[6]  = (__bf16)f1.z; a[7]  = (__bf16)f1.w;
            a[8]  = (__bf16)g0.x; a[9]  = (__bf16)g0.y;
            a[10] = (__bf16)g0.z; a[11] = (__bf16)g0.w;
            a[12] = (__bf16)g1.x; a[13] = (__bf16)g1.y;
            a[14] = (__bf16)g1.z; a[15] = (__bf16)g1.w;
            afrag[ks] = a;
        }
    }

    vq_wait_async();
    __syncthreads();                                // tile 0 resident

    // ---- running min over all K, tracked per (lane, acc-row) ----
    float bestv[8];
    int   besti[8];
    #pragma unroll
    for (int j = 0; j < 8; ++j) { bestv[j] = 3.0e38f; besti[j] = 0; }

    for (int kt = 0; kt < KTILES; ++kt) {
        const char* buf = s_tiles + (kt & 1) * TILE_SZ;
        const char* row = buf + l16 * LDS_ROW;      // this lane's code row

        // Pull the whole tile's B fragments into registers (one ds clause),
        // then run the 8-WMMA chain back-to-back.
        v16bf breg[KSTEPS];
        #pragma unroll
        for (int ks = 0; ks < KSTEPS; ++ks) {
            const int cb0 = ks * 64 + lhalf * 16;   // byte offset in row
            const v8bf blo = *(const v8bf*)(row + cb0);
            const v8bf bhi = *(const v8bf*)(row + cb0 + 32);
            v16bf b;
            #pragma unroll
            for (int i = 0; i < 8; ++i) { b[i] = blo[i]; b[8 + i] = bhi[i]; }
            breg[ks] = b;
        }

        // Overlap: async-fill the other buffer with the next tile.
        if (kt + 1 < KTILES)
            vq_fill_tile(cbb, kt + 1, s_tiles + ((kt + 1) & 1) * TILE_SZ, tid);

        const int   code = kt * 16 + l16;           // this lane's code column
        const float csq  = c_sqr[code];

        v8f acc = {};
        #pragma unroll
        for (int ks = 0; ks < KSTEPS; ++ks)
            acc = __builtin_amdgcn_wmma_f32_16x16x32_bf16(
                      false, afrag[ks], false, breg[ks], (short)0, acc,
                      false, false);

        #pragma unroll
        for (int j = 0; j < 8; ++j) {
            const float score = csq - 2.0f * acc[j];
            if (score < bestv[j]) { bestv[j] = score; besti[j] = code; }
        }

        if (kt + 1 < KTILES) vq_wait_async();
        __syncthreads();                            // next tile ready; this
                                                    // buffer free for refill
    }

    // ---- cross-lane argmin: butterfly within each 16-lane half ----
    #pragma unroll
    for (int off = 8; off >= 1; off >>= 1) {
        #pragma unroll
        for (int j = 0; j < 8; ++j) {
            const float ov = __shfl_xor(bestv[j], off, 32);
            const int   oi = __shfl_xor(besti[j], off, 32);
            if (ov < bestv[j] || (ov == bestv[j] && oi < besti[j])) {
                bestv[j] = ov; besti[j] = oi;
            }
        }
    }
    // lanes 0-15 hold rows mbase+j; lanes 16-31 hold rows mbase+8+j
    if (l16 == 0) {
        #pragma unroll
        for (int j = 0; j < 8; ++j)
            s_idx[wave * 16 + lhalf * 8 + j] = besti[j];
    }
    __syncthreads();

    // ---- gather + write: codes (x2) and indices ----
    float* out_codes0 = out;
    float* out_codes1 = out + (size_t)VQ_N * VQ_D;
    float* out_idx    = out + (size_t)2 * VQ_N * VQ_D;

    const float4* cb4 = (const float4*)cb;
    // 128 rows * 64 float4 = 8192 float4 per WG; 32 per thread
    #pragma unroll 4
    for (int it = 0; it < 32; ++it) {
        const int e    = tid + it * 256;
        const int row  = e >> 6;                 // 0..127
        const int d4   = e & 63;                 // float4 column
        const int code = s_idx[row];
        const float4 v = cb4[(size_t)code * 64 + d4];
        const size_t o = (size_t)(wgrow + row) * 64 + d4;
        ((float4*)out_codes0)[o] = v;
        ((float4*)out_codes1)[o] = v;
    }
    if (tid < ROWS_PER_WG)
        out_idx[wgrow + tid] = (float)s_idx[tid];
}

// ------------------------------- launcher -----------------------------------
extern "C" void kernel_launch(void* const* d_in, const int* in_sizes, int n_in,
                              void* d_out, int out_size, void* d_ws, size_t ws_size,
                              hipStream_t stream) {
    const float* x  = (const float*)d_in[0];   // [B,T,D] f32
    const float* cb = (const float*)d_in[1];   // [K,D]   f32
    float* out = (float*)d_out;

    // workspace: [0,4KB) c_sqr f32[K]; [4KB, 4KB+512KB) bf16 codebook
    float*  c_sqr = (float*)d_ws;
    __bf16* cbb   = (__bf16*)((char*)d_ws + 4096);

    vq_prep_kernel<<<VQ_K, VQ_D, 0, stream>>>(cb, cbb, c_sqr);
    vq_main_kernel<<<VQ_N / ROWS_PER_WG, 256, 0, stream>>>(x, cb, cbb, c_sqr, out);
}
